// Transformer_75874892251800
// MI455X (gfx1250) — compile-verified, matrix operand order checked by reading, not run on hardware
//
#include <hip/hip_runtime.h>
#include <hip/hip_bf16.h>

// ---------------------------------------------------------------------------
// Graph-transformer forward for MI455X (gfx1250, wave32, WMMA).
// All GEMMs + attention contractions use v_wmma_f32_16x16x32_bf16.
// GEMM A-tiles staged with GLOBAL_LOAD_ASYNC_TO_LDS_B128 (ASYNCcnt path).
// ---------------------------------------------------------------------------

#define BB 32
#define LL 256
#define EE 768
#define HH 12
#define DHH 64
#define FFD 3072
#define NNUMD 16
#define NCATD 4

typedef __attribute__((ext_vector_type(16))) __bf16 v16bf;
typedef __attribute__((ext_vector_type(8)))  __bf16 v8bf;
typedef __attribute__((ext_vector_type(8)))  float  v8f;

__device__ __forceinline__ unsigned short f2bf(float f) {
  unsigned int u = __float_as_uint(f);
  unsigned int r = (u + 0x7FFFu + ((u >> 16) & 1u)) >> 16;   // RNE
  return (unsigned short)r;
}

__device__ __forceinline__ v16bf load_frag(const unsigned short* p0,
                                           const unsigned short* p1) {
  union { v16bf v; v8bf h[2]; } u;
  u.h[0] = *(const v8bf*)p0;
  u.h[1] = *(const v8bf*)p1;
  return u.v;
}

// Per-lane async global->LDS copy of 16 bytes (tracked by ASYNCcnt).
__device__ __forceinline__ void async_copy_b128(const void* gptr, const void* lptr) {
  unsigned lds = (unsigned)(unsigned long long)lptr;  // LDS aperture: offset in [31:0]
  asm volatile("global_load_async_to_lds_b128 %0, %1, off"
               :: "v"(lds), "v"(gptr) : "memory");
}
__device__ __forceinline__ void wait_async0() {
  asm volatile("s_wait_asynccnt 0" ::: "memory");
}

__device__ __forceinline__ float rmax16(float v) {
  #pragma unroll
  for (int o = 1; o < 16; o <<= 1) v = fmaxf(v, __shfl_xor(v, o, 32));
  return v;
}
__device__ __forceinline__ float rsum16(float v) {
  #pragma unroll
  for (int o = 1; o < 16; o <<= 1) v += __shfl_xor(v, o, 32);
  return v;
}

// ---------------------------------------------------------------------------
// fp32 -> bf16 conversion
// ---------------------------------------------------------------------------
__global__ void cvt_bf16_kernel(const float* __restrict__ s,
                                unsigned short* __restrict__ d, int n) {
  int i = blockIdx.x * 256 + threadIdx.x;
  if (i < n) d[i] = f2bf(s[i]);
}

// ---------------------------------------------------------------------------
// Embedding: x = num @ w_num + b_num + sum(emb[cat])
// ---------------------------------------------------------------------------
__global__ __launch_bounds__(256)
void embed_kernel(const float* __restrict__ num, const int* __restrict__ cat,
                  const float* __restrict__ w_num, const float* __restrict__ b_num,
                  const float* __restrict__ emb, float* __restrict__ x) {
  int row = blockIdx.x;  // 0..B*L-1
  int c0 = cat[row * NCATD + 0], c1 = cat[row * NCATD + 1];
  int c2 = cat[row * NCATD + 2], c3 = cat[row * NCATD + 3];
  for (int e = threadIdx.x; e < EE; e += 256) {
    float acc = b_num[e];
    #pragma unroll
    for (int j = 0; j < NNUMD; ++j)
      acc += num[row * NNUMD + j] * w_num[j * EE + e];
    acc += emb[c0 * EE + e] + emb[c1 * EE + e] + emb[c2 * EE + e] + emb[c3 * EE + e];
    x[(size_t)row * EE + e] = acc;
  }
}

// ---------------------------------------------------------------------------
// Pairwise log distance
// ---------------------------------------------------------------------------
__global__ __launch_bounds__(256)
void logd_kernel(const float* __restrict__ r, float* __restrict__ logd) {
  int i = blockIdx.x, b = blockIdx.y, j = threadIdx.x;
  const float* ri = r + ((size_t)b * LL + i) * 3;
  const float* rj = r + ((size_t)b * LL + j) * 3;
  float dx = ri[0] - rj[0], dy = ri[1] - rj[1], dz = ri[2] - rj[2];
  float d = sqrtf(dx * dx + dy * dy + dz * dz);
  logd[((size_t)b * LL + i) * LL + j] = (d > 0.f) ? logf(d) : -3.4028235e38f;
}

// ---------------------------------------------------------------------------
// LayerNorm over E, dual fp32 + bf16 output
// ---------------------------------------------------------------------------
__global__ __launch_bounds__(256)
void ln_kernel(const float* __restrict__ x, const float* __restrict__ g,
               const float* __restrict__ bta, float* __restrict__ outF,
               unsigned short* __restrict__ outB) {
  __shared__ float s1[256], s2[256];
  int row = blockIdx.x, t = threadIdx.x;
  const float* xr = x + (size_t)row * EE;
  float v[3];
  float s = 0.f, ss = 0.f;
  #pragma unroll
  for (int j = 0; j < 3; ++j) {
    v[j] = xr[t + j * 256];
    s += v[j];
    ss += v[j] * v[j];
  }
  s1[t] = s; s2[t] = ss;
  __syncthreads();
  for (int o = 128; o > 0; o >>= 1) {
    if (t < o) { s1[t] += s1[t + o]; s2[t] += s2[t + o]; }
    __syncthreads();
  }
  float mean = s1[0] * (1.f / (float)EE);
  float var  = s2[0] * (1.f / (float)EE) - mean * mean;
  float rstd = rsqrtf(var + 1e-5f);
  #pragma unroll
  for (int j = 0; j < 3; ++j) {
    int e = t + j * 256;
    float y = (v[j] - mean) * rstd * g[e] + bta[e];
    if (outF) outF[(size_t)row * EE + e] = y;
    if (outB) outB[(size_t)row * EE + e] = f2bf(y);
  }
}

// ---------------------------------------------------------------------------
// Generic bf16 WMMA GEMM: C[M,N] = A[M,K] @ B[K,N] (+bias,+resid,relu)
// Workgroup tile 64x128, 8 waves each 32x32, K-step 32.
// A-tile: async global->LDS DMA.  B-tile: staged transposed (Bs[n][k]).
// ---------------------------------------------------------------------------
__global__ __launch_bounds__(256)
void gemm_bf16_kernel(const unsigned short* __restrict__ A,
                      const unsigned short* __restrict__ Bw,
                      const float* __restrict__ bias,
                      const float* __restrict__ resid,
                      float* __restrict__ outF,
                      unsigned short* __restrict__ outB,
                      int M, int N, int K, int relu, int qkv) {
  __shared__ __align__(16) unsigned short As[64 * 48];
  __shared__ __align__(16) unsigned short Bs[128 * 48];
  const int tid = threadIdx.x;
  const int wid = tid >> 5, lane = tid & 31, hl = lane >> 4, lr = lane & 15;
  const int wm = wid & 1, wn = wid >> 1;
  const int m0 = blockIdx.y * 64, n0 = blockIdx.x * 128;

  const int am = tid >> 2;        // 0..63
  const int ak = (tid & 3) * 8;   // 0,8,16,24
  const int bk = tid >> 4;        // 0..15
  const int bn = (tid & 15) * 8;  // 0..120

  v8f acc[2][2] = {};

  for (int k0 = 0; k0 < K; k0 += 32) {
    // A tile: per-lane 16B async DMA straight into LDS (no VGPR round-trip).
    async_copy_b128(A + (size_t)(m0 + am) * K + k0 + ak, As + am * 48 + ak);
    // B tile: load row-major, scatter transposed into Bs[n][k].
    union { uint4 u; unsigned short s[8]; } tb0, tb1;
    tb0.u = *(const uint4*)(Bw + (size_t)(k0 + bk) * N + n0 + bn);
    tb1.u = *(const uint4*)(Bw + (size_t)(k0 + bk + 16) * N + n0 + bn);
    #pragma unroll
    for (int j = 0; j < 8; ++j) {
      Bs[(bn + j) * 48 + bk]      = tb0.s[j];
      Bs[(bn + j) * 48 + bk + 16] = tb1.s[j];
    }
    if (k0 + 32 < K) {  // WGP-scope prefetch of next tiles into near caches
      __builtin_prefetch(A + (size_t)(m0 + am) * K + k0 + 32 + ak, 0, 3);
      __builtin_prefetch(Bw + (size_t)(k0 + 32 + bk) * N + n0 + bn, 0, 3);
    }
    wait_async0();
    __syncthreads();

    v16bf af[2], bf[2];
    #pragma unroll
    for (int mt = 0; mt < 2; ++mt) {
      int rw = wm * 32 + mt * 16 + lr;
      af[mt] = load_frag(As + rw * 48 + hl * 8, As + rw * 48 + 16 + hl * 8);
    }
    #pragma unroll
    for (int nt = 0; nt < 2; ++nt) {
      int cl = wn * 32 + nt * 16 + lr;
      bf[nt] = load_frag(Bs + cl * 48 + hl * 8, Bs + cl * 48 + 16 + hl * 8);
    }
    #pragma unroll
    for (int mt = 0; mt < 2; ++mt)
      #pragma unroll
      for (int nt = 0; nt < 2; ++nt)
        acc[mt][nt] = __builtin_amdgcn_wmma_f32_16x16x32_bf16(
            false, af[mt], false, bf[nt], (short)0, acc[mt][nt], false, false);
    __syncthreads();
  }

  // Epilogue
  #pragma unroll
  for (int mt = 0; mt < 2; ++mt) {
    #pragma unroll
    for (int nt = 0; nt < 2; ++nt) {
      int gn = n0 + wn * 32 + nt * 16 + lr;
      #pragma unroll
      for (int rr = 0; rr < 8; ++rr) {
        int gm = m0 + wm * 32 + mt * 16 + hl * 8 + rr;
        float v = acc[mt][nt][rr];
        if (bias)  v += bias[gn];
        if (resid) v += resid[(size_t)gm * N + gn];
        if (relu)  v = fmaxf(v, 0.f);
        if (qkv) {
          int hh = gn >> 6, dd = gn & 63, bb = gm >> 8, ll = gm & 255;
          outB[(((size_t)bb * HH + hh) * LL + ll) * DHH + dd] = f2bf(v);
        } else {
          if (outB) outB[(size_t)gm * N + gn] = f2bf(v);
          if (outF) outF[(size_t)gm * N + gn] = v;
        }
      }
    }
  }
}

// ---------------------------------------------------------------------------
// Flash attention over [B,H,L,DH] bf16 Q/K/V with mask/bias modes:
//   mode 0: no mask/bias        mode 1: diag mask + coef*logd bias
//   mode 2: graph mask (~adj)   mode 3: hybrid (h<H/2: mode1, else mode2)
// Grid: (B*H, 2). 8 waves x 16 query rows each = 128 rows per WG.
// ---------------------------------------------------------------------------
__global__ __launch_bounds__(256)
void attn_kernel(const unsigned short* __restrict__ Q,
                 const unsigned short* __restrict__ Kk,
                 const unsigned short* __restrict__ V,
                 const float* __restrict__ logd,
                 const unsigned char* __restrict__ adj,
                 unsigned short* __restrict__ out,
                 int mode, float coef) {
  __shared__ __align__(16) unsigned short Vt[64 * 48];      // Vt[d][k_local]
  __shared__ __align__(16) unsigned short Pt[8 * 16 * 48];  // per-wave P[m][k]
  const int tid = threadIdx.x;
  const int wid = tid >> 5, lane = tid & 31, hl = lane >> 4, lr = lane & 15;
  const int bh = blockIdx.x;
  const int b = bh / HH, h = bh % HH;
  const int q0 = blockIdx.y * 128 + wid * 16;
  const size_t base = (size_t)bh * LL * DHH;

  // Q fragments (16 rows x 64, two K=32 fragments)
  v16bf qf[2];
  #pragma unroll
  for (int kc = 0; kc < 2; ++kc) {
    const unsigned short* p = Q + base + (size_t)(q0 + lr) * DHH + kc * 32 + hl * 8;
    qf[kc] = load_frag(p, p + 16);
  }

  float mi[8], li[8];
  v8f o[4] = {};
  #pragma unroll
  for (int rr = 0; rr < 8; ++rr) { mi[rr] = -1e30f; li[rr] = 0.f; }

  const int diagmode  = (mode == 1) || (mode == 3 && h < HH / 2);
  const int graphmode = (mode == 2) || (mode == 3 && h >= HH / 2);

  for (int c = 0; c < LL / 32; ++c) {
    __syncthreads();
    {  // cooperatively stage V chunk transposed into LDS
      int key = tid >> 3, dblk = (tid & 7) * 8;
      union { uint4 u; unsigned short s[8]; } tv;
      tv.u = *(const uint4*)(V + base + (size_t)(c * 32 + key) * DHH + dblk);
      #pragma unroll
      for (int j = 0; j < 8; ++j) Vt[(dblk + j) * 48 + key] = tv.s[j];
    }
    __syncthreads();

    // S = Q K^T for 32 keys -> two 16x16 fp32 tiles
    v8f s[2] = {};
    #pragma unroll
    for (int nt = 0; nt < 2; ++nt) {
      int key = c * 32 + nt * 16 + lr;
      #pragma unroll
      for (int kc = 0; kc < 2; ++kc) {
        const unsigned short* p = Kk + base + (size_t)key * DHH + kc * 32 + hl * 8;
        v16bf kf = load_frag(p, p + 16);
        s[nt] = __builtin_amdgcn_wmma_f32_16x16x32_bf16(
            false, qf[kc], false, kf, (short)0, s[nt], false, false);
      }
    }

    // scale + bias + mask
    float sv[2][8];
    #pragma unroll
    for (int nt = 0; nt < 2; ++nt) {
      int key = c * 32 + nt * 16 + lr;
      #pragma unroll
      for (int rr = 0; rr < 8; ++rr) {
        int q = q0 + rr + hl * 8;
        float v = s[nt][rr] * 0.125f;  // 1/sqrt(64)
        if (diagmode) {
          v = (q == key) ? -1e30f
                         : v + coef * logd[((size_t)b * LL + q) * LL + key];
        } else if (graphmode) {
          v = adj[((size_t)b * LL + q) * LL + key] ? v : -1e30f;
        }
        sv[nt][rr] = v;
      }
    }

    // online softmax (row stats via 16-lane shuffles), write P to LDS
    #pragma unroll
    for (int rr = 0; rr < 8; ++rr) {
      float mc = rmax16(fmaxf(sv[0][rr], sv[1][rr]));
      float mnew = fmaxf(mi[rr], mc);
      float sc = __expf(mi[rr] - mnew);
      float p0 = (sv[0][rr] <= -1e29f) ? 0.f : __expf(sv[0][rr] - mnew);
      float p1 = (sv[1][rr] <= -1e29f) ? 0.f : __expf(sv[1][rr] - mnew);
      float rs = rsum16(p0 + p1);
      li[rr] = li[rr] * sc + rs;
      mi[rr] = mnew;
      #pragma unroll
      for (int dt = 0; dt < 4; ++dt) o[dt][rr] *= sc;
      int prow = rr + hl * 8;
      Pt[wid * 16 * 48 + prow * 48 + lr]      = f2bf(p0);
      Pt[wid * 16 * 48 + prow * 48 + 16 + lr] = f2bf(p1);
    }
    __builtin_amdgcn_wave_barrier();
    asm volatile("s_wait_dscnt 0" ::: "memory");

    // O += P V
    const unsigned short* pp = Pt + wid * 16 * 48 + lr * 48 + hl * 8;
    v16bf pf = load_frag(pp, pp + 16);
    #pragma unroll
    for (int dt = 0; dt < 4; ++dt) {
      const unsigned short* vp = Vt + (dt * 16 + lr) * 48 + hl * 8;
      v16bf vf = load_frag(vp, vp + 16);
      o[dt] = __builtin_amdgcn_wmma_f32_16x16x32_bf16(
          false, pf, false, vf, (short)0, o[dt], false, false);
    }
  }

  // normalize + store to [B*L, E] bf16 (e = h*DH + d)
  #pragma unroll
  for (int dt = 0; dt < 4; ++dt)
    #pragma unroll
    for (int rr = 0; rr < 8; ++rr) {
      int q = q0 + rr + hl * 8;
      float val = o[dt][rr] / li[rr];
      out[((size_t)b * LL + q) * EE + h * DHH + dt * 16 + lr] = f2bf(val);
    }
}

// ---------------------------------------------------------------------------
// Final: mean over L of LN(x), then @ w_out + b_out  -> out[B]
// ---------------------------------------------------------------------------
__global__ __launch_bounds__(256)
void pool_kernel(const float* __restrict__ xln, const float* __restrict__ w_out,
                 const float* __restrict__ b_out, float* __restrict__ out) {
  __shared__ float sm[256];
  int b = blockIdx.x, t = threadIdx.x;
  float acc = 0.f;
  for (int i = t; i < LL * EE; i += 256)
    acc += xln[(size_t)b * LL * EE + i] * w_out[i % EE];
  sm[t] = acc;
  __syncthreads();
  for (int o = 128; o > 0; o >>= 1) {
    if (t < o) sm[t] += sm[t + o];
    __syncthreads();
  }
  if (t == 0) out[b] = sm[0] * (1.f / (float)LL) + b_out[0];
}

// ---------------------------------------------------------------------------
// Host orchestration
// ---------------------------------------------------------------------------
extern "C" void kernel_launch(void* const* d_in, const int* in_sizes, int n_in,
                              void* d_out, int out_size, void* d_ws, size_t ws_size,
                              hipStream_t stream) {
  if (n_in < 108) return;  // expected 5 tensors + 103 param leaves (jax sorted order)

  const float*         nodes_num = (const float*)d_in[0];
  const int*           nodes_cat = (const int*)d_in[1];
  const unsigned char* adj       = (const unsigned char*)d_in[2];
  // d_in[3] = padding (all false) unused
  const float* r      = (const float*)d_in[4];
  const float* b_num  = (const float*)d_in[5];
  const float* b_out  = (const float*)d_in[6];
  // blocks: 6 x 16 leaves in sorted-key order
  enum { I_B1 = 0, I_B2, I_BK, I_BO, I_BQ, I_BV, I_LN1B, I_LN1G,
         I_LN2B, I_LN2G, I_W1, I_W2, I_WK, I_WO, I_WQ, I_WV };
  #define BLK(i, j) ((const float*)d_in[7 + (i) * 16 + (j)])
  const float* emb    = (const float*)d_in[103];
  const float* ln_b   = (const float*)d_in[104];
  const float* ln_g   = (const float*)d_in[105];
  const float* w_num  = (const float*)d_in[106];
  const float* w_out  = (const float*)d_in[107];
  float* out = (float*)d_out;

  // ---- workspace layout ----
  char* ws = (char*)d_ws;
  size_t off = 0;
  auto take = [&](size_t bytes) -> char* {
    char* p = ws + off;
    off += (bytes + 255) & ~(size_t)255;
    return p;
  };
  const size_t ME = (size_t)BB * LL * EE;        // 6291456
  const size_t MF = (size_t)BB * LL * FFD;       // 25165824
  float*          xf   = (float*)take(ME * 4);
  float*          x0f  = (float*)take(ME * 4);
  float*          x2f  = (float*)take(ME * 4);
  unsigned short* lnb  = (unsigned short*)take(ME * 2);
  unsigned short* qb   = (unsigned short*)take(ME * 2);
  unsigned short* kb   = (unsigned short*)take(ME * 2);
  unsigned short* vb   = (unsigned short*)take(ME * 2);
  unsigned short* ab   = (unsigned short*)take(ME * 2);
  unsigned short* h1b  = (unsigned short*)take(MF * 2);
  float*          logd = (float*)take((size_t)BB * LL * LL * 4);
  const size_t WQN = (size_t)EE * EE;            // 589824
  const size_t W1N = (size_t)EE * FFD;           // 2359296
  const size_t BLKW = 4 * WQN + 2 * W1N;         // 7077888 elems per block
  unsigned short* wbf = (unsigned short*)take((size_t)6 * BLKW * 2);
  (void)ws_size;

  auto cvt = [&](const float* src, unsigned short* dst, size_t n) {
    cvt_bf16_kernel<<<dim3((unsigned)((n + 255) / 256)), 256, 0, stream>>>(src, dst, (int)n);
  };

  // ---- convert GEMM weights to bf16 once per launch ----
  for (int i = 0; i < 6; ++i) {
    unsigned short* wb = wbf + (size_t)i * BLKW;
    cvt(BLK(i, I_WQ), wb + 0 * WQN, WQN);
    cvt(BLK(i, I_WK), wb + 1 * WQN, WQN);
    cvt(BLK(i, I_WV), wb + 2 * WQN, WQN);
    cvt(BLK(i, I_WO), wb + 3 * WQN, WQN);
    cvt(BLK(i, I_W1), wb + 4 * WQN, W1N);
    cvt(BLK(i, I_W2), wb + 4 * WQN + W1N, W1N);
  }

  // ---- precompute pairwise log-distance & embedding ----
  logd_kernel<<<dim3(LL, BB), 256, 0, stream>>>(r, logd);
  embed_kernel<<<dim3(BB * LL), 256, 0, stream>>>(nodes_num, nodes_cat, w_num, b_num, emb, xf);

  const dim3 gE(EE / 128, BB * LL / 64);   // GEMM grid for N=768
  const dim3 gF(FFD / 128, BB * LL / 64);  // GEMM grid for N=3072
  const dim3 gLN(BB * LL);
  const dim3 gAT(BB * HH, 2);

  // STACK = "GCHLGF"
  const int   modes[6] = {0, 1, 3, 2, 0, 1};
  const float coefs[6] = {0.f, -2.f, -2.f, 0.f, 0.f, -5.f};

  for (int i = 0; i < 6; ++i) {
    unsigned short* wb = wbf + (size_t)i * BLKW;
    // x0 = LN1(x)  (fp32 for residual, bf16 for GEMMs)
    ln_kernel<<<gLN, 256, 0, stream>>>(xf, BLK(i, I_LN1G), BLK(i, I_LN1B), x0f, lnb);
    // Q/K/V projections -> bf16 [B,H,L,DH]
    gemm_bf16_kernel<<<gE, 256, 0, stream>>>(lnb, wb + 0 * WQN, BLK(i, I_BQ),
                                             nullptr, nullptr, qb, BB * LL, EE, EE, 0, 1);
    gemm_bf16_kernel<<<gE, 256, 0, stream>>>(lnb, wb + 1 * WQN, BLK(i, I_BK),
                                             nullptr, nullptr, kb, BB * LL, EE, EE, 0, 1);
    gemm_bf16_kernel<<<gE, 256, 0, stream>>>(lnb, wb + 2 * WQN, BLK(i, I_BV),
                                             nullptr, nullptr, vb, BB * LL, EE, EE, 0, 1);
    // attention
    attn_kernel<<<gAT, 256, 0, stream>>>(qb, kb, vb, logd, adj, ab, modes[i], coefs[i]);
    // x2 = attn @ wo + bo + x0
    gemm_bf16_kernel<<<gE, 256, 0, stream>>>(ab, wb + 3 * WQN, BLK(i, I_BO),
                                             x0f, x2f, nullptr, BB * LL, EE, EE, 0, 0);
    // x2ln = LN2(x2)  (fp32 -> x0f, bf16 -> lnb)
    ln_kernel<<<gLN, 256, 0, stream>>>(x2f, BLK(i, I_LN2G), BLK(i, I_LN2B), x0f, lnb);
    // h1 = relu(x2ln @ w1 + b1)
    gemm_bf16_kernel<<<gF, 256, 0, stream>>>(lnb, wb + 4 * WQN, BLK(i, I_B1),
                                             nullptr, nullptr, h1b, BB * LL, FFD, EE, 1, 0);
    // x = h1 @ w2 + b2 + x2ln
    gemm_bf16_kernel<<<gE, 256, 0, stream>>>(h1b, wb + 4 * WQN + W1N, BLK(i, I_B2),
                                             x0f, xf, nullptr, BB * LL, EE, FFD, 0, 0);
  }

  // final LN, mean over L, output projection
  ln_kernel<<<gLN, 256, 0, stream>>>(xf, ln_g, ln_b, x0f, lnb);
  pool_kernel<<<dim3(BB), 256, 0, stream>>>(x0f, w_out, b_out, out);
  (void)in_sizes; (void)out_size;
}
